// Net_82764019793921
// MI455X (gfx1250) — compile-verified, matrix operand order checked by reading, not run on hardware
//
#include <hip/hip_runtime.h>
#include <hip/hip_bf16.h>

typedef float v8f __attribute__((ext_vector_type(8)));
typedef __bf16 bf16_t;
typedef bf16_t v16bf __attribute__((ext_vector_type(16)));
typedef bf16_t v8bf __attribute__((ext_vector_type(8)));

union Frag16 { v16bf v; v8bf h[2]; };

#define WAVE_LDS_SYNC() asm volatile("s_wait_dscnt 0x0" ::: "memory")

// ---------------------------------------------------------------------------
// Fused gather + GEMM + scatter for a GCN conv layer.
// Per wave: 32 edges (2 A-tiles, so each B fragment feeds 2 WMMAs) x 64 cols
// (50 valid). A gathered from [h[row] | h[col] | e] with branchless address
// selection (clamped row + v_cndmask zeroing), staged f32->bf16 in LDS.
// f32 accumulation via v_wmma_f32_16x16x32_bf16. Writes message m (+bias) and
// atomically scatter-adds into the node accumulator.
// ---------------------------------------------------------------------------
__global__ __launch_bounds__(128) void conv_edge_gemm(
    const float* __restrict__ h, int hdim,
    const float* __restrict__ esrc, int edim, int estride,
    const int* __restrict__ rowi, const int* __restrict__ coli,
    const bf16_t* __restrict__ wt, int Kdim, int Kpad,
    const float* __restrict__ bias,
    float* __restrict__ mout, int mstride, int mcoloff,
    float* __restrict__ hacc, long E)
{
    __shared__ bf16_t ldsA[4][2][16 * 32];
    const int wave = threadIdx.x >> 5;
    const int lane = threadIdx.x & 31;
    const long ebase = (long)blockIdx.x * 128 + (long)wave * 32;
    const int half = lane >> 4, nn = lane & 15;
    const int r2 = lane >> 1, cb = (lane & 1) * 16;

    long estc[2]; bool ev[2]; int ri[2], ci[2];
#pragma unroll
    for (int s = 0; s < 2; ++s) {
        const long est = ebase + s * 16 + r2;
        ev[s] = est < E;
        estc[s] = ev[s] ? est : (E - 1);       // clamped: always a safe address
        ri[s] = rowi[estc[s]];
        ci[s] = coli[estc[s]];
    }

    v8f acc[2][4] = {};
    const int nk = Kpad >> 5;
    for (int kc = 0; kc < nk; ++kc) {
        const int k0 = kc * 32 + cb;
#pragma unroll
        for (int s = 0; s < 2; ++s) {
#pragma unroll
            for (int j = 0; j < 16; ++j) {
                const int k = k0 + j;
                // branchless source select; all candidate addresses valid
                const float* p;
                if (k < hdim)           p = h + (long)ri[s] * hdim + k;
                else if (k < 2 * hdim)  p = h + (long)ci[s] * hdim + (k - hdim);
                else if (k < Kdim)      p = esrc + estc[s] * (long)estride + (k - 2 * hdim);
                else                    p = h;
                float v = *p;
                v = (ev[s] && k < Kdim) ? v : 0.f;
                ldsA[wave][s][r2 * 32 + cb + j] = (bf16_t)v;
            }
        }
        WAVE_LDS_SYNC();
        Frag16 a[2];
#pragma unroll
        for (int s = 0; s < 2; ++s) {
            a[s].h[0] = *(const v8bf*)(&ldsA[wave][s][nn * 32 + half * 8]);
            a[s].h[1] = *(const v8bf*)(&ldsA[wave][s][nn * 32 + 16 + half * 8]);
        }
        const bf16_t* wb = wt + (long)kc * 32 + half * 8;
#pragma unroll
        for (int t = 0; t < 4; ++t) {
            Frag16 b;
            const bf16_t* wc = wb + (long)(t * 16 + nn) * Kpad;
            b.h[0] = *(const v8bf*)(wc);
            b.h[1] = *(const v8bf*)(wc + 16);
            acc[0][t] = __builtin_amdgcn_wmma_f32_16x16x32_bf16(
                false, a[0].v, false, b.v, (short)0, acc[0][t], false, false);
            acc[1][t] = __builtin_amdgcn_wmma_f32_16x16x32_bf16(
                false, a[1].v, false, b.v, (short)0, acc[1][t], false, false);
        }
        WAVE_LDS_SYNC();
    }

#pragma unroll
    for (int s = 0; s < 2; ++s) {
        const long er = ebase + s * 16 + half * 8;
#pragma unroll
        for (int t = 0; t < 4; ++t) {
            const int col = t * 16 + nn;
            if (col >= 50) continue;
            const float bc = bias[col];
#pragma unroll
            for (int v = 0; v < 8; ++v) {
                const long e = er + v;
                if (e < E) {
                    const float val = acc[s][t][v] + bc;
                    mout[e * (long)mstride + mcoloff + col] = val;
                    atomicAdd(&hacc[(long)rowi[e] * 50 + col], val);
                }
            }
        }
    }
}

// ---------------------------------------------------------------------------
// Dense row-major GEMM: out[rows x ncols] = src[rows x Kdim] @ W + bias.
// W is pre-transposed/padded bf16 [Npad][Kpad]. 32 rows per wave (2 A-tiles
// sharing each B fragment), branchless staging.
// ---------------------------------------------------------------------------
__global__ __launch_bounds__(128) void dense_gemm(
    const float* __restrict__ src, int lda, int Kdim,
    const bf16_t* __restrict__ wt, int Kpad,
    const float* __restrict__ bias, int ncols,
    float* __restrict__ out, int ldo, long rows)
{
    __shared__ bf16_t ldsA[4][2][16 * 32];
    const int wave = threadIdx.x >> 5;
    const int lane = threadIdx.x & 31;
    const long rbase = (long)blockIdx.x * 128 + (long)wave * 32;
    const int coltile = blockIdx.y * 64;
    const int half = lane >> 4, nn = lane & 15;
    const int r2 = lane >> 1, cb = (lane & 1) * 16;

    const float* srow[2]; bool rv[2];
#pragma unroll
    for (int s = 0; s < 2; ++s) {
        const long rr = rbase + s * 16 + r2;
        rv[s] = rr < rows;
        const long rrc = rv[s] ? rr : (rows - 1);
        srow[s] = src + rrc * (long)lda;
    }

    v8f acc[2][4] = {};
    const int nk = Kpad >> 5;
    for (int kc = 0; kc < nk; ++kc) {
        const int k0 = kc * 32 + cb;
#pragma unroll
        for (int s = 0; s < 2; ++s) {
#pragma unroll
            for (int j = 0; j < 16; ++j) {
                const int k = k0 + j;
                const float* p = (k < Kdim) ? (srow[s] + k) : srow[s];
                float v = *p;
                v = (rv[s] && k < Kdim) ? v : 0.f;
                ldsA[wave][s][r2 * 32 + cb + j] = (bf16_t)v;
            }
        }
        WAVE_LDS_SYNC();
        Frag16 a[2];
#pragma unroll
        for (int s = 0; s < 2; ++s) {
            a[s].h[0] = *(const v8bf*)(&ldsA[wave][s][nn * 32 + half * 8]);
            a[s].h[1] = *(const v8bf*)(&ldsA[wave][s][nn * 32 + 16 + half * 8]);
        }
        const bf16_t* wb = wt + (long)kc * 32 + half * 8;
#pragma unroll
        for (int t = 0; t < 4; ++t) {
            Frag16 b;
            const bf16_t* wc = wb + (long)(coltile + t * 16 + nn) * Kpad;
            b.h[0] = *(const v8bf*)(wc);
            b.h[1] = *(const v8bf*)(wc + 16);
            acc[0][t] = __builtin_amdgcn_wmma_f32_16x16x32_bf16(
                false, a[0].v, false, b.v, (short)0, acc[0][t], false, false);
            acc[1][t] = __builtin_amdgcn_wmma_f32_16x16x32_bf16(
                false, a[1].v, false, b.v, (short)0, acc[1][t], false, false);
        }
        WAVE_LDS_SYNC();
    }

#pragma unroll
    for (int s = 0; s < 2; ++s) {
        const long er = rbase + s * 16 + half * 8;
#pragma unroll
        for (int t = 0; t < 4; ++t) {
            const int col = coltile + t * 16 + nn;
            if (col >= ncols) continue;
            const float bc = bias ? bias[col] : 0.f;
#pragma unroll
            for (int v = 0; v < 8; ++v) {
                const long r = er + v;
                if (r < rows) out[r * (long)ldo + col] = acc[s][t][v] + bc;
            }
        }
    }
}

// ---------------------------------------------------------------------------
// Weight prep: f32 (K x Ncol) row-major -> bf16 transposed padded [Npad][Kpad]
// ---------------------------------------------------------------------------
__global__ void prep_w(const float* __restrict__ src, int K, int Ncol,
                       int Kpad, int Npad, bf16_t* __restrict__ dst)
{
    const long i = (long)blockIdx.x * blockDim.x + threadIdx.x;
    if (i >= (long)Npad * Kpad) return;
    const int n = (int)(i / Kpad), k = (int)(i % Kpad);
    dst[i] = (n < Ncol && k < K) ? (bf16_t)src[(long)k * Ncol + n] : (bf16_t)0.f;
}

// ---------------------------------------------------------------------------
// BatchNorm (training-mode batch statistics) helpers
// ---------------------------------------------------------------------------
__global__ void bn_stats(const float* __restrict__ src, long rows, int stride,
                         int cols, float* __restrict__ stats, int chunks)
{
    __shared__ float sh[256], sh2[256];
    const int c = blockIdx.y;
    const long per = (rows + chunks - 1) / chunks;
    long r0 = (long)blockIdx.x * per;
    long r1 = r0 + per; if (r1 > rows) r1 = rows;
    float s = 0.f, s2 = 0.f;
    for (long r = r0 + threadIdx.x; r < r1; r += blockDim.x) {
        const float v = src[r * (long)stride + c];
        s += v; s2 += v * v;
    }
    sh[threadIdx.x] = s; sh2[threadIdx.x] = s2;
    __syncthreads();
    for (int d = 128; d > 0; d >>= 1) {
        if ((int)threadIdx.x < d) {
            sh[threadIdx.x] += sh[threadIdx.x + d];
            sh2[threadIdx.x] += sh2[threadIdx.x + d];
        }
        __syncthreads();
    }
    if (threadIdx.x == 0) {
        atomicAdd(&stats[c], sh[0]);
        atomicAdd(&stats[cols + c], sh2[0]);
    }
}

__global__ void bn_apply(const float* __restrict__ src, int sstride,
                         float* __restrict__ dst, int dstride, int dcoloff,
                         const float* __restrict__ stats,
                         const float* __restrict__ g, const float* __restrict__ beta,
                         long rows, int cols, int elu)
{
    const long i = (long)blockIdx.x * blockDim.x + threadIdx.x;
    if (i >= rows * (long)cols) return;
    const long r = i / cols; const int c = (int)(i % cols);
    const float inv = 1.f / (float)rows;
    const float mu = stats[c] * inv;
    const float var = stats[cols + c] * inv - mu * mu;
    float y = g[c] * (src[r * (long)sstride + c] - mu) * rsqrtf(var + 1e-5f) + beta[c];
    if (elu && y < 0.f) y = __expf(y) - 1.f;
    dst[r * (long)dstride + dcoloff + c] = y;
}

__global__ void zero_f32(float* __restrict__ p, long n)
{
    long i = (long)blockIdx.x * blockDim.x + threadIdx.x;
    const long st = (long)gridDim.x * blockDim.x;
    for (; i < n; i += st) p[i] = 0.f;
}

__global__ void zero_strided(float* __restrict__ p, long rows, int cols, int stride)
{
    const long i = (long)blockIdx.x * blockDim.x + threadIdx.x;
    if (i >= rows * (long)cols) return;
    p[(i / cols) * (long)stride + (i % cols)] = 0.f;
}

// mnu[e, oo] = sum_c cond[col[e], c] * ew[e, c*20+oo]; scatter-add to o[row[e], 250+oo]
__global__ void cond_edge(const float* __restrict__ cond, const float* __restrict__ ew,
                          const int* __restrict__ rowi, const int* __restrict__ coli,
                          long e0, long cnt, float* __restrict__ o)
{
    const long i = (long)blockIdx.x * blockDim.x + threadIdx.x;
    if (i >= cnt * 20) return;
    const long el = i / 20; const int oo = (int)(i % 20);
    const long e = e0 + el;
    const float* cv = cond + (long)coli[e] * 20;
    const float* w = ew + el * 400 + oo;
    float s = 0.f;
#pragma unroll
    for (int c = 0; c < 20; ++c) s += cv[c] * w[c * 20];
    atomicAdd(o + (long)rowi[e] * 300 + 250 + oo, s);
}

// Triangle path with the 50->200 linear fused in (avoids materializing E x 200):
// c_xy[t,oo] = sum_j hh2[edge,j] * (sum_c cv[c] * W2[j, c*10+oo]) + sum_c cv[c]*b2[c*10+oo]
__global__ void tri_kernel(const float* __restrict__ cond, const float* __restrict__ hh2,
                           const float* __restrict__ W2, const float* __restrict__ b2,
                           const int* __restrict__ tn, const int* __restrict__ te,
                           long T, float* __restrict__ o)
{
    const long i = (long)blockIdx.x * blockDim.x + threadIdx.x;
    if (i >= T * 40) return;
    const long t = i / 40; const int q = (int)(i % 40);
    const int grp = q / 10, oo = q % 10;
    const int node = (grp == 1 || grp == 3) ? tn[2 * T + t] : tn[t]; // ck : ci
    const long e = te[(long)grp * T + t]; // ed_ij, ed_kj, ed_ik, ed_ki
    float cvr[20];
    const float* cv = cond + (long)node * 20;
#pragma unroll
    for (int c = 0; c < 20; ++c) cvr[c] = cv[c];
    const float* hr = hh2 + e * 50;
    float val = 0.f;
    for (int j = 0; j < 50; ++j) {
        const float* wr = W2 + j * 200 + oo;
        float ws = 0.f;
#pragma unroll
        for (int c = 0; c < 20; ++c) ws += cvr[c] * wr[c * 10];
        val += hr[j] * ws;
    }
    float bs = 0.f;
#pragma unroll
    for (int c = 0; c < 20; ++c) bs += cvr[c] * b2[c * 10 + oo];
    val += bs;
    const float coef = (grp >= 2) ? 0.5f : 1.f;
    const int dcol = (grp == 0) ? 270 + oo : (grp == 1) ? 280 + oo : 290 + oo;
    atomicAdd(o + (long)tn[T + t] * 300 + dcol, coef * val);
}

// ---------------------------------------------------------------------------
extern "C" void kernel_launch(void* const* d_in, const int* in_sizes, int n_in,
                              void* d_out, int out_size, void* d_ws, size_t ws_size,
                              hipStream_t stream)
{
    (void)in_sizes; (void)n_in; (void)out_size; (void)ws_size;
    const long N = 50000, E = 300000, T = 300000;

    const float* x     = (const float*)d_in[0];
    const int*   ei    = (const int*)d_in[1];
    const float* eattr = (const float*)d_in[2];
    const float* cond  = (const float*)d_in[3];
    const int*   tn    = (const int*)d_in[4];
    const int*   te    = (const int*)d_in[5];
    const int* rowi = ei, * coli = ei + E;

    // params (jax pytree leaf order: dict keys sorted)
    int pi = 6;
    const float *convW[15], *convB[15], *convBeta[15], *convG[15];
    {
        int li = 0;
        for (int t = 0; t < 5; ++t)
            for (int l = 0; l <= t; ++l, ++li) {
                convW[li]    = (const float*)d_in[pi + 0];
                convB[li]    = (const float*)d_in[pi + 1];
                convBeta[li] = (const float*)d_in[pi + 2];
                convG[li]    = (const float*)d_in[pi + 3];
                pi += 4;
            }
    }
    // edge_mlp: W1, W2, b1, b2, beta, g
    const float* m1W1 = (const float*)d_in[pi++]; const float* m1W2 = (const float*)d_in[pi++];
    const float* m1b1 = (const float*)d_in[pi++]; const float* m1b2 = (const float*)d_in[pi++];
    const float* m1beta = (const float*)d_in[pi++]; const float* m1g = (const float*)d_in[pi++];
    const float* m2W1 = (const float*)d_in[pi++]; const float* m2W2 = (const float*)d_in[pi++];
    const float* m2b1 = (const float*)d_in[pi++]; const float* m2b2 = (const float*)d_in[pi++];
    const float* m2beta = (const float*)d_in[pi++]; const float* m2g = (const float*)d_in[pi++];
    // lin / linR: W, b, beta, g
    const float* lW = (const float*)d_in[pi++]; const float* lb = (const float*)d_in[pi++];
    const float* lbeta = (const float*)d_in[pi++]; const float* lg = (const float*)d_in[pi++];
    const float* rW = (const float*)d_in[pi++]; const float* rb = (const float*)d_in[pi++];
    const float* rbeta = (const float*)d_in[pi++]; const float* rg = (const float*)d_in[pi++];

    // outputs: o1 (N,20), o2 (N,196), o (N,300), e_cat (E,250)
    float* o1   = (float*)d_out;
    float* o2   = o1 + N * 20;
    float* oo   = o2 + N * 196;
    float* ecat = oo + N * 300;

    // workspace layout (~135 MB)
    float* wsf   = (float*)d_ws;
    float* stats = wsf;                 // 1024
    float* hA    = wsf + 1024;          // node accum (N*50)
    float* hB    = hA + N * 50;         // BN output  (N*50)
    float* eA    = hB + N * 50;         // edge buffer (E*50), in-place reuse
    float* big   = eA + E * 50;         // 13.2M floats: ew chunks / final pre-acts
    bf16_t* wtb  = (bf16_t*)(big + 13200000L);

    const dim3 B256(256);
    auto gz = [](long n) { return dim3((unsigned)((n + 255) / 256)); };

    // ---- weight prep: f32 -> transposed, padded bf16 ----
    bf16_t* wp = wtb;
    bf16_t* wt_conv[15];
    {
        int li = 0;
        for (int t = 0; t < 5; ++t)
            for (int l = 0; l <= t; ++l, ++li) {
                const int Kd = (l == 0) ? 397 : 150, Kp = (l == 0) ? 416 : 160;
                wt_conv[li] = wp;
                prep_w<<<gz(64L * Kp), B256, 0, stream>>>(convW[li], Kd, 50, Kp, 64, wp);
                wp += 64L * Kp;
            }
    }
    bf16_t* wt_m1W1 = wp; prep_w<<<gz(64L * 256), B256, 0, stream>>>(m1W1, 250, 50, 256, 64, wp);  wp += 64L * 256;
    bf16_t* wt_m1W2 = wp; prep_w<<<gz(448L * 64), B256, 0, stream>>>(m1W2, 50, 400, 64, 448, wp);  wp += 448L * 64;
    bf16_t* wt_m2W1 = wp; prep_w<<<gz(64L * 256), B256, 0, stream>>>(m2W1, 250, 50, 256, 64, wp);  wp += 64L * 256;
    bf16_t* wt_lW   = wp; prep_w<<<gz(64L * 320), B256, 0, stream>>>(lW, 300, 20, 320, 64, wp);    wp += 64L * 320;
    bf16_t* wt_rW   = wp; prep_w<<<gz(256L * 320), B256, 0, stream>>>(rW, 300, 196, 320, 256, wp); wp += 256L * 320;

    // ---- conv towers ----
    {
        int li = 0;
        for (int t = 0; t < 5; ++t) {
            const float* hsrc = x; int hdim = 26;
            const float* esrc = eattr; int edim = 345, estride = 345;
            for (int l = 0; l <= t; ++l, ++li) {
                const bool last = (l == t);
                const int Kdim = 2 * hdim + edim;
                const int Kpad = (l == 0) ? 416 : 160;
                zero_f32<<<dim3(2048), B256, 0, stream>>>(hA, N * 50);
                float* mout; int mstride, mcoloff;
                if (last) { mout = ecat; mstride = 250; mcoloff = t * 50; }
                else      { mout = eA;   mstride = 50;  mcoloff = 0; }
                conv_edge_gemm<<<dim3((unsigned)((E + 127) / 128)), dim3(128), 0, stream>>>(
                    hsrc, hdim, esrc, edim, estride, rowi, coli,
                    wt_conv[li], Kdim, Kpad, convB[li],
                    mout, mstride, mcoloff, hA, E);
                zero_f32<<<dim3(2), B256, 0, stream>>>(stats, 512);
                bn_stats<<<dim3(64, 50), B256, 0, stream>>>(hA, N, 50, 50, stats, 64);
                if (last)
                    bn_apply<<<gz(N * 50), B256, 0, stream>>>(hA, 50, oo, 300, t * 50,
                                                              stats, convG[li], convBeta[li], N, 50, 1);
                else
                    bn_apply<<<gz(N * 50), B256, 0, stream>>>(hA, 50, hB, 50, 0,
                                                              stats, convG[li], convBeta[li], N, 50, 1);
                hsrc = hB; hdim = 50;
                esrc = eA; edim = 50; estride = 50;   // in-place (row read-before-write)
            }
        }
    }

    // ---- edge MLP 1: hh1 = elu(bn(e_cat @ W1 + b1)) ----
    dense_gemm<<<dim3((unsigned)((E + 127) / 128), 1), dim3(128), 0, stream>>>(
        ecat, 250, 250, wt_m1W1, 256, m1b1, 50, eA, 50, E);
    zero_f32<<<dim3(2), B256, 0, stream>>>(stats, 512);
    bn_stats<<<dim3(128, 50), B256, 0, stream>>>(eA, E, 50, 50, stats, 128);
    bn_apply<<<gz(E * 50), B256, 0, stream>>>(eA, 50, eA, 50, 0, stats, m1g, m1beta, E, 50, 1);

    // zero cond_out / cond_tri slice of o, then scatter into it
    zero_strided<<<gz(N * 50), B256, 0, stream>>>(oo + 250, N, 50, 300);

    // ew = hh1 @ W2 + b2 in 32K-edge chunks, consumed immediately
    const long CH = 32768;
    for (long c0 = 0; c0 < E; c0 += CH) {
        const long cnt = (E - c0 < CH) ? (E - c0) : CH;
        dense_gemm<<<dim3((unsigned)((cnt + 127) / 128), 7), dim3(128), 0, stream>>>(
            eA + c0 * 50, 50, 50, wt_m1W2, 64, m1b2, 400, big, 400, cnt);
        cond_edge<<<gz(cnt * 20), B256, 0, stream>>>(cond, big, rowi, coli, c0, cnt, oo);
    }

    // ---- edge MLP 2 (first linear only; second fused into tri_kernel) ----
    dense_gemm<<<dim3((unsigned)((E + 127) / 128), 1), dim3(128), 0, stream>>>(
        ecat, 250, 250, wt_m2W1, 256, m2b1, 50, eA, 50, E);
    zero_f32<<<dim3(2), B256, 0, stream>>>(stats, 512);
    bn_stats<<<dim3(128, 50), B256, 0, stream>>>(eA, E, 50, 50, stats, 128);
    bn_apply<<<gz(E * 50), B256, 0, stream>>>(eA, 50, eA, 50, 0, stats, m2g, m2beta, E, 50, 1);

    tri_kernel<<<gz(T * 40), B256, 0, stream>>>(cond, eA, m2W2, m2b2, tn, te, T, oo);

    // ---- final linears + BN ----
    dense_gemm<<<dim3((unsigned)((N + 127) / 128), 1), dim3(128), 0, stream>>>(
        oo, 300, 300, wt_lW, 320, lb, 20, big, 20, N);
    zero_f32<<<dim3(2), B256, 0, stream>>>(stats, 512);
    bn_stats<<<dim3(64, 20), B256, 0, stream>>>(big, N, 20, 20, stats, 64);
    bn_apply<<<gz(N * 20), B256, 0, stream>>>(big, 20, o1, 20, 0, stats, lg, lbeta, N, 20, 0);

    dense_gemm<<<dim3((unsigned)((N + 127) / 128), 4), dim3(128), 0, stream>>>(
        oo, 300, 300, wt_rW, 320, rb, 196, big, 196, N);
    zero_f32<<<dim3(2), B256, 0, stream>>>(stats, 512);
    bn_stats<<<dim3(64, 196), B256, 0, stream>>>(big, N, 196, 196, stats, 64);
    bn_apply<<<gz(N * 196), B256, 0, stream>>>(big, 196, o2, 196, 0, stats, rg, rbeta, N, 196, 0);
}
